// MarlinLinear_58128087384233
// MI455X (gfx1250) — matrix operand order repaired and checked by reading, not compiled, not run on hardware
//
#include <hip/hip_runtime.h>

// ---- CDNA5 WMMA types -------------------------------------------------------
typedef __bf16 bf16_t;
typedef bf16_t v16bf  __attribute__((ext_vector_type(16)));
typedef bf16_t bf16x2 __attribute__((ext_vector_type(2)));
typedef float  v8f    __attribute__((ext_vector_type(8)));

#define BM 128
#define BN 256
#define BK 64
#define LDK 72   // BK + 8 bf16 pad to stagger LDS banks

// two f32 -> packed bf16 (RTNE) ; should lower to v_cvt_pk_bf16_f32
__device__ __forceinline__ unsigned pkbf(float x, float y) {
    bf16x2 v = { (bf16_t)x, (bf16_t)y };
    return __builtin_bit_cast(unsigned, v);
}

// bits = 0x4300|q pairs == (128+q_lo, 128+q_hi) as bf16x2.
// (128+q) + (-136) is exact in bf16 (both integers <= 8 bits), then one
// rounding multiply by the scale -> v_pk_add_bf16 + v_pk_mul_bf16.
__device__ __forceinline__ unsigned dq2(unsigned bits, bf16x2 sc2) {
    const bf16x2 m136 = __builtin_bit_cast(bf16x2, 0xC308C308u); // (-136,-136)
    bf16x2 q = __builtin_bit_cast(bf16x2, bits);
    bf16x2 r = (q + m136) * sc2;
    return __builtin_bit_cast(unsigned, r);
}

__global__ __launch_bounds__(256)
void marlin_w4_bf16_wmma(const float* __restrict__ A,
                         const int*   __restrict__ B,
                         const float* __restrict__ s,
                         const float* __restrict__ bias,
                         float*       __restrict__ C,
                         int M, int N, int K)
{
    // NOTE: both LDS tiles hold k in the permuted order 0,4,1,5,2,6,3,7 per
    // 8-group (applied identically to A and B, so the WMMA dot product is
    // unchanged). This makes the int4 nibble extraction one AND per pair.
    __shared__ __align__(16) unsigned short Alds[BM][LDK];   // [m][k'] bf16
    __shared__ __align__(16) unsigned short Blds[BN][LDK];   // [n][k'] bf16

    const int tid  = threadIdx.x;
    const int lane = tid & 31;
    const int wave = tid >> 5;
    const int waveM = wave & 1;    // 2 waves along M -> 64 rows each
    const int waveN = wave >> 1;   // 4 waves along N -> 64 cols each

    const int nBase = blockIdx.x * BN;
    const int mBase = blockIdx.y * BM;

    // A staging: thread owns one 8-column group of one row; 4 row passes
    const int ag8  = tid & 7;      // 8-col group: cols ag8*8 .. ag8*8+7
    const int arow = tid >> 3;     // 0..31 (+32 per pass)

    const int nkt = K / BK;

    float4 aReg[4][2];
    int2   bReg[4];
    float  sReg;

    // -------- stage: global -> registers --------
    auto load_stage = [&](int kt) {
        const int k0 = kt * BK;
#pragma unroll
        for (int p = 0; p < 4; ++p) {
            const int row = p * 32 + arow;
            const float* base = A + (size_t)(mBase + row) * K + k0 + ag8 * 8;
            aReg[p][0] = *(const float4*)(base);
            aReg[p][1] = *(const float4*)(base + 4);
        }
        const int tBase = k0 >> 4;   // packed-tile row base
#pragma unroll
        for (int p = 0; p < 4; ++p)
            bReg[p] = *(const int2*)(B + (size_t)(tBase + p) * (2 * N)
                                       + 2 * (nBase + tid));
        // one group scale covers all 4 tile rows of this BK=64 stage
        sReg = s[(size_t)(k0 >> 7) * N + (nBase + tid)];
    };

    // -------- stage: registers -> LDS (convert / dequantize) --------
    auto store_stage = [&]() {
        // A: pos 2j <- col j, pos 2j+1 <- col j+4  (the k' permutation)
#pragma unroll
        for (int p = 0; p < 4; ++p) {
            const float4 a = aReg[p][0], b = aReg[p][1];
            uint4 pk = make_uint4(pkbf(a.x, b.x), pkbf(a.y, b.y),
                                  pkbf(a.z, b.z), pkbf(a.w, b.w));
            *(uint4*)&Alds[p * 32 + arow][ag8 * 8] = pk;
        }
        // B: AND-mask extraction yields the same k' permutation for free
        const bf16x2 sc2 = { (bf16_t)sReg, (bf16_t)sReg };
#pragma unroll
        for (int p = 0; p < 4; ++p) {
            const unsigned w0 = (unsigned)bReg[p].x;
            const unsigned w1 = (unsigned)bReg[p].y;
            uint4 lo, hi;
            lo.x = dq2(0x43004300u | ( w0        & 0x000F000Fu), sc2);
            lo.y = dq2(0x43004300u | ((w0 >>  4) & 0x000F000Fu), sc2);
            lo.z = dq2(0x43004300u | ((w0 >>  8) & 0x000F000Fu), sc2);
            lo.w = dq2(0x43004300u | ((w0 >> 12) & 0x000F000Fu), sc2);
            hi.x = dq2(0x43004300u | ( w1        & 0x000F000Fu), sc2);
            hi.y = dq2(0x43004300u | ((w1 >>  4) & 0x000F000Fu), sc2);
            hi.z = dq2(0x43004300u | ((w1 >>  8) & 0x000F000Fu), sc2);
            hi.w = dq2(0x43004300u | ((w1 >> 12) & 0x000F000Fu), sc2);
            *(uint4*)&Blds[tid][p * 16]     = lo;
            *(uint4*)&Blds[tid][p * 16 + 8] = hi;
        }
    };

    v8f acc[4][4];
#pragma unroll
    for (int mt = 0; mt < 4; ++mt)
#pragma unroll
        for (int nt = 0; nt < 4; ++nt)
            acc[mt][nt] = (v8f){0.f,0.f,0.f,0.f,0.f,0.f,0.f,0.f};

    const int hi8 = (lane >= 16) ? 8 : 0;   // A frag lane-half K offset
    const int l15 = lane & 15;

    union Frag { uint4 u[2]; v16bf v; };

    // -------- compute one BK stage from LDS --------
    auto compute_stage = [&]() {
#pragma unroll
        for (int kk = 0; kk < BK; kk += 32) {
            Frag fa[4], fb[4];
            // A 16x32 frag: lanes 0-15 -> K0-7,16-23 ; lanes 16-31 -> K8-15,24-31
#pragma unroll
            for (int mt = 0; mt < 4; ++mt) {
                const int row = waveM * 64 + mt * 16 + l15;
                fa[mt].u[0] = *(const uint4*)&Alds[row][kk + hi8];
                fa[mt].u[1] = *(const uint4*)&Alds[row][kk + 16 + hi8];
            }
            // B 32x16 frag: lanes 0-15 -> K0-7,16-23 ; lanes 16-31 -> K8-15,24-31
            const int kb = kk + hi8;
#pragma unroll
            for (int nt = 0; nt < 4; ++nt) {
                const int col = waveN * 64 + nt * 16 + l15;
                fb[nt].u[0] = *(const uint4*)&Blds[col][kb];
                fb[nt].u[1] = *(const uint4*)&Blds[col][kb + 16];
            }
#pragma unroll
            for (int mt = 0; mt < 4; ++mt)
#pragma unroll
                for (int nt = 0; nt < 4; ++nt)
                    acc[mt][nt] = __builtin_amdgcn_wmma_f32_16x16x32_bf16(
                        false, fa[mt].v, false, fb[nt].v,
                        (short)0, acc[mt][nt], false, false);
        }
    };

    // -------- software-pipelined main loop --------
    load_stage(0);
    for (int kt = 0; kt < nkt; ++kt) {
        __syncthreads();            // previous stage's LDS reads complete
        store_stage();
        __syncthreads();            // stage visible to all waves
        if (kt + 1 < nkt) {
            load_stage(kt + 1);     // overlap next global loads with WMMA
            if (kt + 2 < nkt) {     // L2 prefetch two stages ahead
                const int k2 = (kt + 2) * BK;
                __builtin_prefetch(A + (size_t)(mBase + arow) * K + k2 + ag8 * 8, 0, 0);
                __builtin_prefetch(B + (size_t)(k2 >> 4) * (2 * N)
                                     + 2 * (nBase + tid), 0, 0);
            }
        }
        compute_stage();
    }

    // -------- epilogue: C = acc + bias --------
#pragma unroll
    for (int mt = 0; mt < 4; ++mt) {
#pragma unroll
        for (int nt = 0; nt < 4; ++nt) {
            const int col = nBase + waveN * 64 + nt * 16 + l15;
            const float bv = bias[col];
#pragma unroll
            for (int i = 0; i < 8; ++i) {
                const int row = mBase + waveM * 64 + mt * 16 + i + hi8;
                C[(size_t)row * N + col] = acc[mt][nt][i] + bv;
            }
        }
    }
}

extern "C" void kernel_launch(void* const* d_in, const int* in_sizes, int n_in,
                              void* d_out, int out_size, void* d_ws, size_t ws_size,
                              hipStream_t stream) {
    const float* A    = (const float*)d_in[0];
    const int*   B    = (const int*)  d_in[1];
    const float* s    = (const float*)d_in[2];
    const float* bias = (const float*)d_in[3];
    float* C = (float*)d_out;

    const int N = in_sizes[3];                                // out_features
    const int K = (int)(((long long)in_sizes[2] / N) * 128);  // s is [K/128, N]
    const int M = (int)((long long)in_sizes[0] / K);          // A is [M, K]

    dim3 grid(N / BN, M / BM);
    marlin_w4_bf16_wmma<<<grid, dim3(256), 0, stream>>>(A, B, s, bias, C, M, N, K);
}
